// VectorQuantizer_63986422776172
// MI455X (gfx1250) — matrix operand order repaired and checked by reading, not compiled, not run on hardware
//
#include <hip/hip_runtime.h>

typedef __attribute__((ext_vector_type(16))) __bf16 v16bf;
typedef __attribute__((ext_vector_type(4)))  __bf16 v4bf;
typedef __attribute__((ext_vector_type(8)))  float  v8f;

#define N_TOK 32768
#define K_CB  8192
#define D_DIM 768

#define BM     128   // token rows per block
#define BK     128   // codebook cols per block
#define BD     32    // D-chunk per WMMA k-step
#define LDSROW 40    // padded row stride in bf16 elems (80B, 16B-aligned, bank-conflict-free)

// wave tiling: 8 waves arranged 4 (M) x 2 (N); each wave = 32 rows x 64 cols
//            = 2 M-tiles x 4 N-tiles -> 6 fragment-pair loads for 24 WMMAs/step

// ---------- helpers ----------

// split f32 into hi bf16 + residual bf16 using native RTNE converters,
// store 4 packed halves with one ds_store_b64 each
__device__ __forceinline__ void split_store(unsigned short* hiP, unsigned short* loP,
                                            float4 v, float scale) {
    float y0 = scale * v.x, y1 = scale * v.y, y2 = scale * v.z, y3 = scale * v.w;
    __bf16 h0 = (__bf16)y0, h1 = (__bf16)y1, h2 = (__bf16)y2, h3 = (__bf16)y3;
    __bf16 l0 = (__bf16)(y0 - (float)h0);
    __bf16 l1 = (__bf16)(y1 - (float)h1);
    __bf16 l2 = (__bf16)(y2 - (float)h2);
    __bf16 l3 = (__bf16)(y3 - (float)h3);
    v4bf hv = { h0, h1, h2, h3 };
    v4bf lv = { l0, l1, l2, l3 };
    *(v4bf*)hiP = hv;
    *(v4bf*)loP = lv;
}

// load a 32-byte WMMA fragment (two ds_load_b128) from LDS
__device__ __forceinline__ v16bf ld_frag(const unsigned short* p0, const unsigned short* p1) {
    union { uint4 q[2]; v16bf v; } u;
    u.q[0] = *(const uint4*)p0;
    u.q[1] = *(const uint4*)p1;
    return u.v;
}

// ---------- kernel 0: init packed (key|idx) min buffer ----------

__global__ void vq_init_packed(unsigned long long* __restrict__ packed) {
    int i = blockIdx.x * blockDim.x + threadIdx.x;
    if (i < N_TOK) packed[i] = ~0ull;
}

// ---------- kernel 1: e_sq[k] = sum_d embed[k][d]^2 (one wave per code) ----------

__global__ __launch_bounds__(256) void vq_esq(const float* __restrict__ E,
                                              float* __restrict__ esq) {
    const int lane = threadIdx.x & 31;
    const int wave = threadIdx.x >> 5;
    const int k = blockIdx.x * 8 + wave;
    float s = 0.0f;
    #pragma unroll
    for (int j = 0; j < D_DIM / 32; ++j) {
        float v = E[(size_t)k * D_DIM + j * 32 + lane];
        s = fmaf(v, v, s);
    }
    #pragma unroll
    for (int m = 1; m < 32; m <<= 1) s += __shfl_xor(s, m, 32);
    if (lane == 0) esq[k] = s;
}

// ---------- kernel 2: tiled split-bf16 WMMA GEMM + online argmin ----------

__global__ __launch_bounds__(256) void vq_gemm_argmin(
    const float* __restrict__ X, const float* __restrict__ E,
    const float* __restrict__ esq, unsigned long long* __restrict__ packed) {

    __shared__ __align__(16) unsigned short sXhi[BM * LDSROW];
    __shared__ __align__(16) unsigned short sXlo[BM * LDSROW];
    __shared__ __align__(16) unsigned short sEhi[BK * LDSROW];
    __shared__ __align__(16) unsigned short sElo[BK * LDSROW];
    __shared__ float sEsq[BK];

    const int tid  = threadIdx.x;
    const int lane = tid & 31;
    const int wave = tid >> 5;
    const int codeBase = blockIdx.x * BK;
    const int rowBase  = blockIdx.y * BM;

    if (tid < BK) sEsq[tid] = esq[codeBase + tid];

    const v8f vzero = {0.f, 0.f, 0.f, 0.f, 0.f, 0.f, 0.f, 0.f};
    v8f acc[2][4];
    #pragma unroll
    for (int mi = 0; mi < 2; ++mi)
        #pragma unroll
        for (int ni = 0; ni < 4; ++ni) acc[mi][ni] = vzero;

    float4 xv[4], ev[4];

    auto load_stage = [&](int d0) {
        #pragma unroll
        for (int i = 0; i < 4; ++i) {
            int f  = tid + 256 * i;           // float4 slot: 128 rows x 8 float4
            int r  = f >> 3;
            int c4 = f & 7;
            xv[i] = *(const float4*)(X + (size_t)(rowBase  + r) * D_DIM + d0 + c4 * 4);
            ev[i] = *(const float4*)(E + (size_t)(codeBase + r) * D_DIM + d0 + c4 * 4);
        }
    };

    auto store_stage = [&]() {
        #pragma unroll
        for (int i = 0; i < 4; ++i) {
            int f = tid + 256 * i;
            int r = f >> 3;
            int c = (f & 7) * 4;
            // fold the exact -2 factor into X's bf16 split
            split_store(&sXhi[r * LDSROW + c], &sXlo[r * LDSROW + c], xv[i], -2.0f);
            split_store(&sEhi[r * LDSROW + c], &sElo[r * LDSROW + c], ev[i],  1.0f);
        }
    };

    // WMMA fragment addressing (ISA 16-bit A/B layouts, wave32)
    const int lrow    = lane & 15;
    const int halfSel = lane >> 4;              // 0: lanes 0-15, 1: lanes 16-31
    const int waveM   = wave >> 1;              // 0..3  -> 32-row group
    const int waveN   = wave & 1;               // 0..1  -> 64-col group
    const int aE0     = halfSel ? 8 : 0;        // A: K{0..7,16..23} vs K{8..15,24..31}
    const int bE0     = halfSel ? 8 : 0;        // B: K{0..7,16..23} vs K{8..15,24..31}

    load_stage(0);

    #pragma unroll 1
    for (int s = 0; s < D_DIM / BD; ++s) {
        store_stage();
        __syncthreads();
        if (s + 1 < D_DIM / BD) load_stage((s + 1) * BD);

        v16bf aH[2], aL[2];
        #pragma unroll
        for (int mi = 0; mi < 2; ++mi) {
            const int aRow = waveM * 32 + mi * 16 + lrow;
            const unsigned short* aHb = &sXhi[aRow * LDSROW + aE0];
            const unsigned short* aLb = &sXlo[aRow * LDSROW + aE0];
            aH[mi] = ld_frag(aHb, aHb + 16);
            aL[mi] = ld_frag(aLb, aLb + 16);
        }

        #pragma unroll
        for (int ni = 0; ni < 4; ++ni) {
            const int bRow = waveN * 64 + ni * 16 + lrow;
            const unsigned short* bHb = &sEhi[bRow * LDSROW + bE0];
            const unsigned short* bLb = &sElo[bRow * LDSROW + bE0];
            v16bf bH = ld_frag(bHb, bHb + 16);
            v16bf bL = ld_frag(bLb, bLb + 16);
            #pragma unroll
            for (int mi = 0; mi < 2; ++mi) {
                // acc += (-2x)_h*e_h + (-2x)_h*e_l + (-2x)_l*e_h  (~fp32 accuracy)
                acc[mi][ni] = __builtin_amdgcn_wmma_f32_16x16x32_bf16(
                    false, aH[mi], false, bH, (short)0, acc[mi][ni], false, false);
                acc[mi][ni] = __builtin_amdgcn_wmma_f32_16x16x32_bf16(
                    false, aH[mi], false, bL, (short)0, acc[mi][ni], false, false);
                acc[mi][ni] = __builtin_amdgcn_wmma_f32_16x16x32_bf16(
                    false, aL[mi], false, bH, (short)0, acc[mi][ni], false, false);
            }
        }
        __syncthreads();
    }

    // epilogue: dist = e_sq[k] - 2 x.e ; per-row min over this wave's 64 codes,
    // then one global_atomic_min_u64 per row on (sortable_key<<32 | index).
    #pragma unroll
    for (int mi = 0; mi < 2; ++mi) {
        #pragma unroll
        for (int i = 0; i < 8; ++i) {
            float bestV = __uint_as_float(0x7F7FFFFFu);   // +FLT_MAX
            int   bestK = 0x7FFFFFFF;
            #pragma unroll
            for (int ni = 0; ni < 4; ++ni) {
                int   kc = waveN * 64 + ni * 16 + lrow;
                float v  = sEsq[kc] + acc[mi][ni][i];
                int   k  = codeBase + kc;
                if (v < bestV || (v == bestV && k < bestK)) { bestV = v; bestK = k; }
            }
            // reduce across each 16-lane half (C/D layout: halves hold different rows)
            #pragma unroll
            for (int m = 1; m < 16; m <<= 1) {
                float ov = __shfl_xor(bestV, m, 32);
                int   ok = __shfl_xor(bestK, m, 32);
                if (ov < bestV || (ov == bestV && ok < bestK)) { bestV = ov; bestK = ok; }
            }
            if (lrow == 0) {
                int row = rowBase + waveM * 32 + mi * 16 + i + halfSel * 8;
                unsigned int fb  = __float_as_uint(bestV);
                unsigned int key = fb ^ ((fb >> 31) ? 0xFFFFFFFFu : 0x80000000u);
                unsigned long long p = ((unsigned long long)key << 32) | (unsigned int)bestK;
                atomicMin(&packed[row], p);
            }
        }
    }
}

// ---------- kernel 3: gather codebook rows + emit indices ----------

__global__ __launch_bounds__(256) void vq_gather(const float* __restrict__ E,
                                                 const unsigned long long* __restrict__ packed,
                                                 float* __restrict__ out) {
    const int row = blockIdx.x;
    const unsigned int idx = (unsigned int)(packed[row] & 0xFFFFFFFFu);
    const float* src = E + (size_t)idx * D_DIM;
    float* dst = out + (size_t)row * D_DIM;
    #pragma unroll
    for (int j = threadIdx.x; j < D_DIM; j += 256) dst[j] = src[j];
    if (threadIdx.x == 0) out[(size_t)N_TOK * D_DIM + row] = (float)idx;
}

// ---------- launch ----------

extern "C" void kernel_launch(void* const* d_in, const int* in_sizes, int n_in,
                              void* d_out, int out_size, void* d_ws, size_t ws_size,
                              hipStream_t stream) {
    (void)in_sizes; (void)n_in; (void)out_size; (void)ws_size;

    const float* X = (const float*)d_in[0];   // [N, D] f32
    const float* E = (const float*)d_in[1];   // [K, D] f32
    float* out = (float*)d_out;               // [N*D] quantize, then [N] indices

    float* esq = (float*)d_ws;                                             // K floats (32 KB)
    unsigned long long* packed =
        (unsigned long long*)((char*)d_ws + (size_t)K_CB * sizeof(float)); // N u64 (256 KB)

    vq_init_packed<<<dim3(N_TOK / 256), dim3(256), 0, stream>>>(packed);
    vq_esq<<<dim3(K_CB / 8), dim3(256), 0, stream>>>(E, esq);
    vq_gemm_argmin<<<dim3(K_CB / BK, N_TOK / BM), dim3(256), 0, stream>>>(X, E, esq, packed);
    vq_gather<<<dim3(N_TOK), dim3(256), 0, stream>>>(E, packed, out);
}
